// AttentionBlock_61246233641678
// MI455X (gfx1250) — matrix operand order repaired and checked by reading, not compiled, not run on hardware
//
#include <hip/hip_runtime.h>
#include <hip/hip_bf16.h>
#include <math.h>

// ---------------- types / constants ----------------
typedef __bf16 bf16;
typedef bf16  v16bf __attribute__((ext_vector_type(16)));
typedef float v8f   __attribute__((ext_vector_type(8)));
typedef unsigned int u32x4 __attribute__((ext_vector_type(4)));
typedef int i32x4 __attribute__((ext_vector_type(4)));
typedef int i32x8 __attribute__((ext_vector_type(8)));

#define AS1 __attribute__((address_space(1)))
#define AS3 __attribute__((address_space(3)))

#define WMMA_BF16(a, b, c) \
    __builtin_amdgcn_wmma_f32_16x16x32_bf16(false, (a), false, (b), (short)0, (c), false, false)

#if defined(__AMDGCN__) && __has_builtin(__builtin_amdgcn_global_load_async_to_lds_b128) && \
    __has_builtin(__builtin_amdgcn_s_wait_asynccnt)
#define HAS_ASYNC 1
#else
#define HAS_ASYNC 0
#endif

#if defined(__AMDGCN__) && __has_builtin(__builtin_amdgcn_tensor_load_to_lds) && \
    __has_builtin(__builtin_amdgcn_s_wait_tensorcnt)
#define HAS_TDM 1
#else
#define HAS_TDM 0
#endif

static constexpr int kB = 8;      // batch
static constexpr int kC = 256;    // channels
static constexpr int kN = 4096;   // H*W
static constexpr int kG = 16;     // groups
static constexpr int kCG = kC / kG;
static constexpr float kScale = 0.0625f;  // C^-0.5 = 1/16

// ---------------- async / TDM helpers ----------------
#if HAS_ASYNC
static __device__ inline void async_b128(const void* gsrc, void* lds) {
    __builtin_amdgcn_global_load_async_to_lds_b128(
        (AS1 i32x4*)(gsrc), (AS3 i32x4*)(lds), 0, 0);
}
#endif

#if HAS_TDM
// 2D TDM load: tile_d0 (units of 8B) x tile_d1 rows, row stride stride0 (8B units).
// D# bit packing per CDNA5 ISA sec 8.3/8.4 (data_size=3 -> 8 bytes).
static __device__ inline void tdm_load(unsigned lds_off, const void* gsrc,
                                       unsigned tile_d0, unsigned tile_d1,
                                       unsigned tensor_d0, unsigned tensor_d1,
                                       unsigned long long stride0) {
    unsigned long long ga = (unsigned long long)gsrc;
    u32x4 g0;
    g0[0] = 1u;                                   // count=1, user descriptor
    g0[1] = lds_off;                              // lds_addr
    g0[2] = (unsigned)ga;                         // global_addr[31:0]
    g0[3] = (unsigned)((ga >> 32) & 0x01FFFFFFu)  // global_addr[56:32]
            | (2u << 30);                         // type=2 ("image")
    i32x8 g1;
    g1[0] = (int)(3u << 16);                                  // data_size=8B
    g1[1] = (int)((tensor_d0 & 0xFFFFu) << 16);               // tensor_dim0 lo
    g1[2] = (int)(((tensor_d0 >> 16) & 0xFFFFu) |
                  ((tensor_d1 & 0xFFFFu) << 16));             // td0 hi | td1 lo
    g1[3] = (int)(((tensor_d1 >> 16) & 0xFFFFu) |
                  ((tile_d0 & 0xFFFFu) << 16));               // td1 hi | tile_dim0
    g1[4] = (int)(tile_d1 & 0xFFFFu);                         // tile_dim1, tile_dim2=0
    g1[5] = (int)(unsigned)(stride0 & 0xFFFFFFFFull);         // dim0_stride lo
    g1[6] = (int)(unsigned)((stride0 >> 32) & 0xFFFFull);     // dim0_stride hi
    g1[7] = 0;
    i32x4 z4; z4[0] = 0; z4[1] = 0; z4[2] = 0; z4[3] = 0;
#if __clang_major__ >= 23
    i32x8 z8;
    #pragma unroll
    for (int i = 0; i < 8; ++i) z8[i] = 0;
    __builtin_amdgcn_tensor_load_to_lds(g0, g1, z4, z4, z8, 0);
#else
    __builtin_amdgcn_tensor_load_to_lds(g0, g1, z4, z4, 0);
#endif
}
#endif

// ---------------- WMMA fragment loaders (LDS, bf16) ----------------
// A operand: 16x32 (MxK). Row M = lane&15. Lanes 0-15: K=0..7 then 16..23;
// lanes 16-31: K=8..15 then 24..31.
static __device__ inline v16bf frag_a(const bf16* p, int stride) {
    int lane = threadIdx.x & 31;
    int r = lane & 15, sel = lane >> 4;
    union { v16bf v; uint4 q[2]; } f;
    const bf16* row = p + r * stride;
    f.q[0] = *(const uint4*)(row + sel * 8);
    f.q[1] = *(const uint4*)(row + 16 + sel * 8);
    return f.v;
}

// B operand: 32x16 (KxN), source layout [n][k] (K contiguous per column).
// Column N = lane&15. Lanes 0-15: K=0..7 then 16..23; lanes 16-31: K=8..15 then 24..31.
static __device__ inline v16bf frag_b(const bf16* p, int stride) {
    int lane = threadIdx.x & 31;
    int r = lane & 15, sel = lane >> 4;
    union { v16bf v; uint4 q[2]; } f;
    const bf16* row = p + r * stride;
    f.q[0] = *(const uint4*)(row + sel * 8);
    f.q[1] = *(const uint4*)(row + 16 + sel * 8);
    return f.v;
}

// ---------------- 1) GroupNorm statistics ----------------
__global__ __launch_bounds__(256) void gn_stats(const float* __restrict__ x,
                                                float* __restrict__ stats) {
    int bg = blockIdx.x;            // b * G + g
    const float* p = x + (size_t)bg * kCG * kN;
    float s = 0.f, ss = 0.f;
    for (int i = threadIdx.x; i < kCG * kN; i += 256) {
        float v = p[i];
        s += v; ss += v * v;
    }
    __shared__ float rs[256], rq[256];
    rs[threadIdx.x] = s; rq[threadIdx.x] = ss;
    __syncthreads();
    for (int o = 128; o > 0; o >>= 1) {
        if (threadIdx.x < o) {
            rs[threadIdx.x] += rs[threadIdx.x + o];
            rq[threadIdx.x] += rq[threadIdx.x + o];
        }
        __syncthreads();
    }
    if (threadIdx.x == 0) {
        float inv_n = 1.0f / (float)(kCG * kN);
        float mu = rs[0] * inv_n;
        float var = rq[0] * inv_n - mu * mu;
        stats[bg * 2 + 0] = mu;
        stats[bg * 2 + 1] = rsqrtf(var + 1e-5f);
    }
}

// ---------------- 2) Normalize + affine, write xnT [B,N,C] bf16 ----------------
__global__ __launch_bounds__(256) void gn_norm(const float* __restrict__ x,
                                               const float* __restrict__ gamma,
                                               const float* __restrict__ beta,
                                               const float* __restrict__ stats,
                                               bf16* __restrict__ xnT) {
    size_t idx = (size_t)blockIdx.x * 256 + threadIdx.x;
    int c = (int)(idx % kC);
    int n = (int)((idx / kC) % kN);
    int b = (int)(idx / ((size_t)kC * kN));
    int g = c >> 4;
    float mu = stats[(b * kG + g) * 2 + 0];
    float rstd = stats[(b * kG + g) * 2 + 1];
    float v = x[((size_t)b * kC + c) * kN + n];
    float yn = (v - mu) * rstd * gamma[c] + beta[c];
    xnT[((size_t)b * kN + n) * kC + c] = (bf16)yn;
}

// ---------------- 3) QKV GEMM ----------------
__global__ __launch_bounds__(256) void qkv_gemm(const float* __restrict__ W,
                                                const float* __restrict__ bias,
                                                const bf16* __restrict__ xnT,
                                                bf16* __restrict__ qT,
                                                bf16* __restrict__ kT,
                                                bf16* __restrict__ vN) {
    int bn = blockIdx.x, bo = blockIdx.y, b = blockIdx.z;
    __shared__ __align__(16) bf16 Wl[64][32];
    __shared__ __align__(16) bf16 Xl[64][32];

    int tid = threadIdx.x;
    int wave = tid >> 5, lane = tid & 31;
    int lane15 = lane & 15, hi = lane >> 4;
    int ot = wave & 3;
    int ntb = (wave >> 2) * 2;

    v8f acc[2] = {};

    int lr = tid >> 2;            // 0..63
    int lc = (tid & 3) * 8;       // 0,8,16,24

    for (int kk = 0; kk < 8; ++kk) {
        const float* wsrc = W + ((size_t)(bo * 64 + lr)) * kC + kk * 32 + lc;
        #pragma unroll
        for (int j = 0; j < 8; ++j) Wl[lr][lc + j] = (bf16)wsrc[j];
        const bf16* xsrc = &xnT[((size_t)b * kN + bn * 64 + lr) * kC + kk * 32 + lc];
#if HAS_ASYNC
        async_b128(xsrc, &Xl[lr][lc]);
        __builtin_amdgcn_s_wait_asynccnt(0);
#else
        *(uint4*)&Xl[lr][lc] = *(const uint4*)xsrc;
#endif
        __syncthreads();

        #pragma unroll
        for (int t = 0; t < 2; ++t) {
            v16bf af = frag_a(&Wl[ot * 16][0], 32);
            v16bf bfr = frag_b(&Xl[(ntb + t) * 16][0], 32);
            acc[t] = WMMA_BF16(af, bfr, acc[t]);
        }
        __syncthreads();
    }

    #pragma unroll
    for (int t = 0; t < 2; ++t) {
        int n = bn * 64 + (ntb + t) * 16 + lane15;
        #pragma unroll
        for (int j = 0; j < 8; ++j) {
            int o = bo * 64 + ot * 16 + j + 8 * hi;
            float val = acc[t][j] + bias[o];
            bf16 bv = (bf16)val;
            if (o < 256)
                qT[((size_t)b * kN + n) * kC + o] = bv;
            else if (o < 512)
                kT[((size_t)b * kN + n) * kC + (o - 256)] = bv;
            else
                vN[((size_t)b * kC + (o - 512)) * kN + n] = bv;
        }
    }
}

// ---------------- 4) Flash attention: grid (N/64, B), 256 threads ----------------
// K/V tiles double-buffered via TDM; s_wait_tensorcnt(2) overlaps next-tile DMA
// with current-tile WMMA work.
__global__ __launch_bounds__(256) void attn_kernel(const bf16* __restrict__ qT,
                                                   const bf16* __restrict__ kT,
                                                   const bf16* __restrict__ vN,
                                                   bf16* __restrict__ hT) {
    int qt = blockIdx.x, b = blockIdx.y;
    __shared__ __align__(16) bf16 Qs[64][256];
    __shared__ __align__(16) bf16 Ks[2][64][256];
    __shared__ __align__(16) bf16 Vs[2][256][64];
    __shared__ __align__(16) bf16 Ps[64][64];
    __shared__ float smax[64], ssum[64], sscale[64];
    __shared__ float pmax[4][64], psum[4][64];

    int tid = threadIdx.x;
    int wave = tid >> 5, lane = tid & 31;
    int lane15 = lane & 15, hi = lane >> 4;
    int mt = wave & 3;
    int ntb = (wave >> 2) * 2;

    // Load query tile (64 x 256 bf16)
    {
        int r = tid >> 2;
        int base = (tid & 3) * 64;
        const bf16* src = &qT[((size_t)b * kN + qt * 64 + r) * kC + base];
        #pragma unroll
        for (int j = 0; j < 8; ++j) {
#if HAS_ASYNC
            async_b128(src + j * 8, &Qs[r][base + j * 8]);
#else
            *(uint4*)&Qs[r][base + j * 8] = *(const uint4*)(src + j * 8);
#endif
        }
#if HAS_ASYNC
        __builtin_amdgcn_s_wait_asynccnt(0);
#endif
    }
    if (tid < 64) { smax[tid] = -3.0e38f; ssum[tid] = 0.f; }

#if HAS_TDM
    // Prologue: kick off K/V pair for tile 0 into buffer 0.
    if (wave == 0) {
        tdm_load((unsigned)(size_t)(AS3 char*)&Ks[0][0][0],
                 &kT[((size_t)b * kN + 0) * kC],
                 4096u, 1u, 4096u, 1u, 4096ull);
        tdm_load((unsigned)(size_t)(AS3 char*)&Vs[0][0][0],
                 &vN[(size_t)b * kC * kN + 0],
                 16u, 256u, 16u, 256u, 1024ull);
    }
#endif

    v8f Oacc[8] = {};   // [i*4 + nt]: c-tile = wave*2+i, n-tile = nt

    for (int kt = 0; kt < 64; ++kt) {
        int cur = kt & 1;
        int nxt = cur ^ 1;
        __syncthreads();   // prior iteration's consumers done with buf[nxt] and Ps
#if HAS_TDM
        if (wave == 0) {
            if (kt + 1 < 64) {
                // Stream next K/V pair into the other buffer while we compute.
                tdm_load((unsigned)(size_t)(AS3 char*)&Ks[nxt][0][0],
                         &kT[((size_t)b * kN + (kt + 1) * 64) * kC],
                         4096u, 1u, 4096u, 1u, 4096ull);
                tdm_load((unsigned)(size_t)(AS3 char*)&Vs[nxt][0][0],
                         &vN[(size_t)b * kC * kN + (kt + 1) * 64],
                         16u, 256u, 16u, 256u, 1024ull);
                // <=2 outstanding  ==>  pair(kt) has landed (in-order per wave)
                __builtin_amdgcn_s_wait_tensorcnt(2);
            } else {
                __builtin_amdgcn_s_wait_tensorcnt(0);
            }
        }
#else
        {
            int r = tid >> 2;
            int base = (tid & 3) * 64;
            const bf16* src = &kT[((size_t)b * kN + kt * 64 + r) * kC + base];
            #pragma unroll
            for (int j = 0; j < 8; ++j)
                *(uint4*)&Ks[cur][r][base + j * 8] = *(const uint4*)(src + j * 8);
        }
        {
            const bf16* src = &vN[((size_t)b * kC + tid) * kN + kt * 64];
            #pragma unroll
            for (int j = 0; j < 8; ++j)
                *(uint4*)&Vs[cur][tid][j * 8] = *(const uint4*)(src + j * 8);
        }
        if (kt + 1 < 64)
            __builtin_prefetch(&kT[((size_t)b * kN + (kt + 1) * 64) * kC], 0, 0);
#endif
        __syncthreads();

        // S^T[m,n] = sum_c K^T[m,c] * Q[c,n], scaled
        v8f S[2] = {};
        #pragma unroll
        for (int cc = 0; cc < 8; ++cc) {
            v16bf af = frag_a(&Ks[cur][mt * 16][0] + cc * 32, 256);
            #pragma unroll
            for (int t = 0; t < 2; ++t) {
                v16bf bfr = frag_b(&Qs[(ntb + t) * 16][0] + cc * 32, 256);
                S[t] = WMMA_BF16(af, bfr, S[t]);
            }
        }
        #pragma unroll
        for (int t = 0; t < 2; ++t)
            #pragma unroll
            for (int j = 0; j < 8; ++j) S[t][j] *= kScale;

        // per-column (n) partial max over this m-tile
        #pragma unroll
        for (int t = 0; t < 2; ++t) {
            float m = S[t][0];
            #pragma unroll
            for (int j = 1; j < 8; ++j) m = fmaxf(m, S[t][j]);
            m = fmaxf(m, __shfl_xor(m, 16, 32));
            if (hi == 0) pmax[mt][(ntb + t) * 16 + lane15] = m;
        }
        __syncthreads();

        if (tid < 64) {
            float nm = smax[tid];
            #pragma unroll
            for (int r = 0; r < 4; ++r) nm = fmaxf(nm, pmax[r][tid]);
            float sc = __expf(smax[tid] - nm);
            sscale[tid] = sc;
            smax[tid] = nm;
            ssum[tid] *= sc;
        }
        __syncthreads();

        // P = exp(S - max), pack to bf16 at Ps[n][m]; column partial sums
        #pragma unroll
        for (int t = 0; t < 2; ++t) {
            int ncol = (ntb + t) * 16 + lane15;
            float mx = smax[ncol];
            float ps = 0.f;
            #pragma unroll
            for (int j = 0; j < 8; ++j) {
                float p = __expf(S[t][j] - mx);
                S[t][j] = p;
                ps += p;
            }
            ps += __shfl_xor(ps, 16, 32);
            if (hi == 0) psum[mt][ncol] = ps;
            #pragma unroll
            for (int j = 0; j < 8; ++j)
                Ps[ncol][mt * 16 + j + 8 * hi] = (bf16)S[t][j];
        }

        // rescale running O accumulators (column-local scale)
        #pragma unroll
        for (int nt = 0; nt < 4; ++nt) {
            float sc = sscale[nt * 16 + lane15];
            #pragma unroll
            for (int i = 0; i < 2; ++i)
                #pragma unroll
                for (int j = 0; j < 8; ++j) Oacc[i * 4 + nt][j] *= sc;
        }
        __syncthreads();

        if (tid < 64) {
            float s = ssum[tid];
            #pragma unroll
            for (int r = 0; r < 4; ++r) s += psum[r][tid];
            ssum[tid] = s;
        }

        // O[c,n] += V[c,m] * P^T[m,n]
        #pragma unroll
        for (int i = 0; i < 2; ++i) {
            int ct = wave * 2 + i;
            #pragma unroll
            for (int half = 0; half < 2; ++half) {
                v16bf af = frag_a(&Vs[cur][ct * 16][0] + half * 32, 64);
                #pragma unroll
                for (int nt = 0; nt < 4; ++nt) {
                    v16bf bfr = frag_b(&Ps[nt * 16][0] + half * 32, 64);
                    Oacc[i * 4 + nt] = WMMA_BF16(af, bfr, Oacc[i * 4 + nt]);
                }
            }
        }
    }
    __syncthreads();

    // finalize: divide by l, write hT [B,N,C] bf16
    #pragma unroll
    for (int nt = 0; nt < 4; ++nt) {
        float inv = 1.0f / ssum[nt * 16 + lane15];
        int n = qt * 64 + nt * 16 + lane15;
        #pragma unroll
        for (int i = 0; i < 2; ++i) {
            #pragma unroll
            for (int j = 0; j < 8; ++j) {
                int c = (wave * 2 + i) * 16 + j + 8 * hi;
                hT[((size_t)b * kN + n) * kC + c] = (bf16)(Oacc[i * 4 + nt][j] * inv);
            }
        }
    }
}

// ---------------- 5) Projection + residual, fp32 out ----------------
__global__ __launch_bounds__(256) void proj_res(const float* __restrict__ W,
                                                const float* __restrict__ bias,
                                                const bf16* __restrict__ hT,
                                                const float* __restrict__ x,
                                                float* __restrict__ out) {
    int bn = blockIdx.x, bo = blockIdx.y, b = blockIdx.z;
    __shared__ __align__(16) bf16 Wl[64][32];
    __shared__ __align__(16) bf16 Xl[64][32];

    int tid = threadIdx.x;
    int wave = tid >> 5, lane = tid & 31;
    int lane15 = lane & 15, hi = lane >> 4;
    int ot = wave & 3;
    int ntb = (wave >> 2) * 2;

    v8f acc[2] = {};
    int lr = tid >> 2;
    int lc = (tid & 3) * 8;

    for (int kk = 0; kk < 8; ++kk) {
        const float* wsrc = W + ((size_t)(bo * 64 + lr)) * kC + kk * 32 + lc;
        #pragma unroll
        for (int j = 0; j < 8; ++j) Wl[lr][lc + j] = (bf16)wsrc[j];
        const bf16* xsrc = &hT[((size_t)b * kN + bn * 64 + lr) * kC + kk * 32 + lc];
#if HAS_ASYNC
        async_b128(xsrc, &Xl[lr][lc]);
        __builtin_amdgcn_s_wait_asynccnt(0);
#else
        *(uint4*)&Xl[lr][lc] = *(const uint4*)xsrc;
#endif
        __syncthreads();

        #pragma unroll
        for (int t = 0; t < 2; ++t) {
            v16bf af = frag_a(&Wl[ot * 16][0], 32);
            v16bf bfr = frag_b(&Xl[(ntb + t) * 16][0], 32);
            acc[t] = WMMA_BF16(af, bfr, acc[t]);
        }
        __syncthreads();
    }

    #pragma unroll
    for (int t = 0; t < 2; ++t) {
        int n = bn * 64 + (ntb + t) * 16 + lane15;
        #pragma unroll
        for (int j = 0; j < 8; ++j) {
            int o = bo * 64 + ot * 16 + j + 8 * hi;
            size_t idx = ((size_t)b * kC + o) * kN + n;
            out[idx] = x[idx] + acc[t][j] + bias[o];
        }
    }
}

// ---------------- host launch ----------------
extern "C" void kernel_launch(void* const* d_in, const int* in_sizes, int n_in,
                              void* d_out, int out_size, void* d_ws, size_t ws_size,
                              hipStream_t stream) {
    const float* x       = (const float*)d_in[0];
    const float* gamma   = (const float*)d_in[1];
    const float* beta    = (const float*)d_in[2];
    const float* qkv_w   = (const float*)d_in[3];
    const float* qkv_b   = (const float*)d_in[4];
    const float* proj_w  = (const float*)d_in[5];
    const float* proj_b  = (const float*)d_in[6];
    float* out = (float*)d_out;

    char* ws = (char*)d_ws;
    float* stats = (float*)ws;                       // 256 floats
    const size_t elems = (size_t)kB * kN * kC;       // 8.4M bf16 per tensor
    bf16* xnT = (bf16*)(ws + 1024);
    bf16* qT  = xnT + elems;
    bf16* kT  = qT + elems;
    bf16* vN  = kT + elems;
    bf16* hT  = xnT;                                 // xnT dead after QKV; reuse

    gn_stats<<<kB * kG, 256, 0, stream>>>(x, stats);
    gn_norm<<<(int)(elems / 256), 256, 0, stream>>>(x, gamma, beta, stats, xnT);
    qkv_gemm<<<dim3(kN / 64, 12, kB), 256, 0, stream>>>(qkv_w, qkv_b, xnT, qT, kT, vN);
    attn_kernel<<<dim3(kN / 64, kB), 256, 0, stream>>>(qT, kT, vN, hT);
    proj_res<<<dim3(kN / 64, 4, kB), 256, 0, stream>>>(proj_w, proj_b, hT, x, out);
}
